// M3GNet_actor_7550552507076
// MI455X (gfx1250) — matrix operand order repaired and checked by reading, not compiled, not run on hardware
//
#include <hip/hip_runtime.h>
#include <hip/hip_bf16.h>
#include <stdint.h>

// ---------------------------------------------------------------------------
// CDNA5 (gfx1250) wave32 WMMA types
// ---------------------------------------------------------------------------
typedef __attribute__((ext_vector_type(16))) __bf16 v16bf;
typedef __attribute__((ext_vector_type(8)))  float  v8f;

union Frag {
  v16bf v;
  uint4 q[2];
  unsigned short h[16];
};

// Native f32 <-> bf16 conversion (clang emits v_cvt_pk_bf16_f32 etc.)
__device__ __forceinline__ unsigned short f2bf(float f) {
  __bf16 b = (__bf16)f;
  return __builtin_bit_cast(unsigned short, b);
}
__device__ __forceinline__ float bf2f(unsigned short h) {
  return (float)__builtin_bit_cast(__bf16, h);
}
// Fast activations: v_exp_f32 + v_rcp_f32 (no IEEE division sequence)
__device__ __forceinline__ float fsig(float x) {
  return __builtin_amdgcn_rcpf(1.0f + __expf(-x));
}
__device__ __forceinline__ float fsilu(float x) { return x * fsig(x); }

__device__ __forceinline__ v8f wmma_bf16(const Frag& a, const Frag& b, v8f c) {
  // (neg_a, A, neg_b, B, c_mod, C, reuse_a, reuse_b)
  return __builtin_amdgcn_wmma_f32_16x16x32_bf16(false, a.v, false, b.v,
                                                 (short)0, c, false, false);
}

// B fragment: weights stored transposed W^T[dout][din], stride multiple of 8.
// lane holds column N = lane&15; K chunks k0+g*8 and k0+16+g*8 (g = lane>>4).
__device__ __forceinline__ Frag load_b(const unsigned short* Wt, int stride,
                                       int ntile, int k0, int lane) {
  Frag f;
  int n = ntile * 16 + (lane & 15);
  int k = k0 + ((lane >> 4) << 3);
  const uint4* p = reinterpret_cast<const uint4*>(Wt + (size_t)n * stride + k);
  f.q[0] = p[0];
  f.q[1] = p[2];
  return f;
}

// A fragment from a bf16 row pointer (global hbf row or LDS activation row).
// lane holds row M = lane&15; K chunks k0+g*8 and k0+16+g*8 (each 8 bf16).
__device__ __forceinline__ Frag load_a(const unsigned short* rowp, int k0, int g) {
  Frag f;
  f.q[0] = *reinterpret_cast<const uint4*>(rowp + k0 + g * 8);
  f.q[1] = *reinterpret_cast<const uint4*>(rowp + k0 + 16 + g * 8);
  return f;
}

#define WPB 4   // waves (16-row tiles) per 128-thread block

// ---------------------------------------------------------------------------
// Weight prep: f32 W[din][dout] -> bf16 W^T[dout][stride] (zero padded)
// ---------------------------------------------------------------------------
__global__ void k_convert(const float* __restrict__ W, unsigned short* __restrict__ Wt,
                          int din, int dout, int stride) {
  int i = blockIdx.x * 256 + threadIdx.x;
  if (i >= dout * stride) return;
  int n = i / stride, k = i - n * stride;
  float v = (k < din) ? W[(size_t)k * dout + n] : 0.0f;
  Wt[i] = f2bf(v);
}

// ---------------------------------------------------------------------------
// Geometry prep (block-invariant): scatter-mean of edge_vec, cos_theta
// ---------------------------------------------------------------------------
__global__ void k_scatter_vec(const long long* __restrict__ ecol,
                              const float* __restrict__ evec,
                              float* __restrict__ vsum, float* __restrict__ vcnt, int E) {
  int e = blockIdx.x * 256 + threadIdx.x;
  if (e >= E) return;
  long long c = ecol[e];
  atomicAdd(&vsum[c * 3 + 0], evec[e * 3 + 0]);
  atomicAdd(&vsum[c * 3 + 1], evec[e * 3 + 1]);
  atomicAdd(&vsum[c * 3 + 2], evec[e * 3 + 2]);
  atomicAdd(&vcnt[c], 1.0f);
}

__global__ void k_div_vec(float* __restrict__ vsum, const float* __restrict__ vcnt, int N) {
  int i = blockIdx.x * 256 + threadIdx.x;
  if (i >= N) return;
  float inv = 1.0f / fmaxf(vcnt[i], 1.0f);
  vsum[i * 3 + 0] *= inv;
  vsum[i * 3 + 1] *= inv;
  vsum[i * 3 + 2] *= inv;
}

__global__ void k_cos(const long long* __restrict__ ecol, const float* __restrict__ evec,
                      const float* __restrict__ elen, const float* __restrict__ nvec,
                      float* __restrict__ ecos, int E) {
  int e = blockIdx.x * 256 + threadIdx.x;
  if (e >= E) return;
  long long c = ecol[e];
  float cx = nvec[c * 3 + 0], cy = nvec[c * 3 + 1], cz = nvec[c * 3 + 2];
  float dot = evec[e * 3 + 0] * cx + evec[e * 3 + 1] * cy + evec[e * 3 + 2] * cz;
  float norm = elen[e] * (sqrtf(cx * cx + cy * cy + cz * cz) + 1e-6f);
  float ct = dot / norm;
  ecos[e] = fminf(1.0f, fmaxf(-1.0f, ct));
}

// ---------------------------------------------------------------------------
// Embedding: h = emb2(silu(emb1([x|fs|fn])))  (WMMA 32->128->128 per 16 nodes)
// ---------------------------------------------------------------------------
__global__ void __launch_bounds__(128) k_embed(
    const float* __restrict__ x, const float* __restrict__ fs, const float* __restrict__ fn,
    const unsigned short* __restrict__ W1, const float* __restrict__ b1,
    const unsigned short* __restrict__ W2, const float* __restrict__ b2,
    float* __restrict__ h, unsigned short* __restrict__ hbf, int N) {
  __shared__ unsigned short sxin[WPB][16 * 32];
  __shared__ unsigned short sact[WPB][16 * 128];
  const int w = threadIdx.x >> 5, lane = threadIdx.x & 31;
  const int tile = blockIdx.x * WPB + w;
  if (tile * 16 >= N) return;
  const int tb = tile * 16, m16 = lane & 15, g = lane >> 4, h8 = g * 8;
  unsigned short* xin = sxin[w];
  unsigned short* act = sact[w];

  // stage 16x32 input tile into LDS as bf16
#pragma unroll
  for (int e = 0; e < 16; ++e) {
    int flat = lane * 16 + e;
    int nl = flat >> 5, feat = flat & 31;
    int node = tb + nl;
    float v;
    if (feat < 16)      v = x[node * 16 + feat];
    else if (feat < 28) v = fs[node * 12 + feat - 16];
    else                v = fn[node * 4 + feat - 28];
    xin[nl * 32 + feat] = f2bf(v);
  }

  v8f acc[8];
#pragma unroll
  for (int nt = 0; nt < 8; ++nt) {
    float bv = b1[nt * 16 + m16];
#pragma unroll
    for (int r = 0; r < 8; ++r) acc[nt][r] = bv;
  }
  {
    Frag A = load_a(xin + m16 * 32, 0, g);
#pragma unroll
    for (int nt = 0; nt < 8; ++nt) {
      Frag B = load_b(W1, 32, nt, 0, lane);
      acc[nt] = wmma_bf16(A, B, acc[nt]);
    }
  }
#pragma unroll
  for (int nt = 0; nt < 8; ++nt)
#pragma unroll
    for (int r = 0; r < 8; ++r)
      act[(r + h8) * 128 + nt * 16 + m16] = f2bf(fsilu(acc[nt][r]));

#pragma unroll
  for (int nt = 0; nt < 8; ++nt) {
    float bv = b2[nt * 16 + m16];
#pragma unroll
    for (int r = 0; r < 8; ++r) acc[nt][r] = bv;
  }
#pragma unroll
  for (int ks = 0; ks < 4; ++ks) {
    Frag A = load_a(act + m16 * 128, ks * 32, g);
#pragma unroll
    for (int nt = 0; nt < 8; ++nt) {
      Frag B = load_b(W2, 128, nt, ks * 32, lane);
      acc[nt] = wmma_bf16(A, B, acc[nt]);
    }
  }
#pragma unroll
  for (int nt = 0; nt < 8; ++nt)
#pragma unroll
    for (int r = 0; r < 8; ++r) {
      size_t idx = (size_t)(tb + r + h8) * 128 + nt * 16 + m16;
      float v = acc[nt][r];
      h[idx] = v;
      hbf[idx] = f2bf(v);
    }
}

// ---------------------------------------------------------------------------
// Fused edge kernel: e1 -> e2 -> t1(+cos) -> t2 -> scatter(m*g) per 16 edges
// ---------------------------------------------------------------------------
__global__ void __launch_bounds__(128) k_edge(
    const unsigned short* __restrict__ hbf,
    const long long* __restrict__ erow, const long long* __restrict__ ecol,
    const float* __restrict__ elen, const float* __restrict__ ecos,
    const unsigned short* __restrict__ We1, const float* __restrict__ be1,
    const unsigned short* __restrict__ We2, const float* __restrict__ be2,
    const unsigned short* __restrict__ Wt1, const float* __restrict__ bt1,
    const unsigned short* __restrict__ Wt2, const float* __restrict__ bt2,
    float* __restrict__ aggr, int E) {
  __shared__ unsigned short sb0[WPB][16 * 128];
  __shared__ unsigned short sb1[WPB][16 * 128];
  const int w = threadIdx.x >> 5, lane = threadIdx.x & 31;
  const int tile = blockIdx.x * WPB + w;
  if (tile * 16 >= E) return;
  const int tb = tile * 16, m16 = lane & 15, g = lane >> 4, h8 = g * 8;
  unsigned short* a0 = sb0[w];
  unsigned short* a1 = sb1[w];

  const long long ri = erow[tb + m16];
  const long long ci = ecol[tb + m16];
  const unsigned short* rowR = hbf + (size_t)ri * 128;
  const unsigned short* rowC = hbf + (size_t)ci * 128;
  const float dlen = elen[tb + m16];

  v8f acc[8];
  // -------- e1 : [h[row] | rbf | h[col]] (K=320) -> 128 --------
#pragma unroll
  for (int nt = 0; nt < 8; ++nt) {
    float bv = be1[nt * 16 + m16];
#pragma unroll
    for (int r = 0; r < 8; ++r) acc[nt][r] = bv;
  }
#pragma unroll
  for (int ks = 0; ks < 10; ++ks) {
    const int k0 = ks * 32;
    Frag A;
    if (k0 < 128) {
      A = load_a(rowR, k0, g);
    } else if (k0 >= 192) {
      A = load_a(rowC, k0 - 192, g);
    } else {
      // RBF bins computed in-register: exp(-((d-c)^2)/w^2), c=5k/63, 1/w^2=163.84
#pragma unroll
      for (int i = 0; i < 16; ++i) {
        int kk = (k0 - 128) + (i < 8 ? h8 + i : 16 + h8 + (i - 8));
        float d = dlen - 5.0f * (float)kk * (1.0f / 63.0f);
        A.h[i] = f2bf(__expf(-d * d * 163.84f));
      }
    }
#pragma unroll
    for (int nt = 0; nt < 8; ++nt) {
      Frag B = load_b(We1, 320, nt, k0, lane);
      acc[nt] = wmma_bf16(A, B, acc[nt]);
    }
  }
#pragma unroll
  for (int nt = 0; nt < 8; ++nt)
#pragma unroll
    for (int r = 0; r < 8; ++r)
      a0[(r + h8) * 128 + nt * 16 + m16] = f2bf(fsilu(acc[nt][r]));

  // -------- e2 : 128 -> 128, m = silu(...) in a1 --------
#pragma unroll
  for (int nt = 0; nt < 8; ++nt) {
    float bv = be2[nt * 16 + m16];
#pragma unroll
    for (int r = 0; r < 8; ++r) acc[nt][r] = bv;
  }
#pragma unroll
  for (int ks = 0; ks < 4; ++ks) {
    Frag A = load_a(a0 + m16 * 128, ks * 32, g);
#pragma unroll
    for (int nt = 0; nt < 8; ++nt) {
      Frag B = load_b(We2, 128, nt, ks * 32, lane);
      acc[nt] = wmma_bf16(A, B, acc[nt]);
    }
  }
#pragma unroll
  for (int nt = 0; nt < 8; ++nt)
#pragma unroll
    for (int r = 0; r < 8; ++r)
      a1[(r + h8) * 128 + nt * 16 + m16] = f2bf(fsilu(acc[nt][r]));

  // -------- t1 : [m | cos] (K=129) -> 128; cos column as VALU rank-1 --------
  float cosr[8];
#pragma unroll
  for (int r = 0; r < 8; ++r) cosr[r] = ecos[tb + r + h8];
#pragma unroll
  for (int nt = 0; nt < 8; ++nt) {
    float bv = bt1[nt * 16 + m16];
    float w129 = bf2f(Wt1[(size_t)(nt * 16 + m16) * 136 + 128]);
#pragma unroll
    for (int r = 0; r < 8; ++r) acc[nt][r] = bv + cosr[r] * w129;
  }
#pragma unroll
  for (int ks = 0; ks < 4; ++ks) {
    Frag A = load_a(a1 + m16 * 128, ks * 32, g);
#pragma unroll
    for (int nt = 0; nt < 8; ++nt) {
      Frag B = load_b(Wt1, 136, nt, ks * 32, lane);
      acc[nt] = wmma_bf16(A, B, acc[nt]);
    }
  }
#pragma unroll
  for (int nt = 0; nt < 8; ++nt)
#pragma unroll
    for (int r = 0; r < 8; ++r)
      a0[(r + h8) * 128 + nt * 16 + m16] = f2bf(fsilu(acc[nt][r]));

  // -------- t2 : 128 -> 128, g = sigmoid(...) --------
#pragma unroll
  for (int nt = 0; nt < 8; ++nt) {
    float bv = bt2[nt * 16 + m16];
#pragma unroll
    for (int r = 0; r < 8; ++r) acc[nt][r] = bv;
  }
#pragma unroll
  for (int ks = 0; ks < 4; ++ks) {
    Frag A = load_a(a0 + m16 * 128, ks * 32, g);
#pragma unroll
    for (int nt = 0; nt < 8; ++nt) {
      Frag B = load_b(Wt2, 128, nt, ks * 32, lane);
      acc[nt] = wmma_bf16(A, B, acc[nt]);
    }
  }

  // -------- scatter: aggr[col] += m * sigmoid(t2) --------
  long long cr[8];
#pragma unroll
  for (int r = 0; r < 8; ++r) cr[r] = ecol[tb + r + h8];
#pragma unroll
  for (int nt = 0; nt < 8; ++nt)
#pragma unroll
    for (int r = 0; r < 8; ++r) {
      float gg = fsig(acc[nt][r]);
      float mv = bf2f(a1[(r + h8) * 128 + nt * 16 + m16]);
      atomicAdd(&aggr[(size_t)cr[r] * 128 + nt * 16 + m16], mv * gg);
    }
}

// ---------------------------------------------------------------------------
// Node update: h += silu(n2(silu(n1([h|aggr]))))  (K=256 -> 128 -> 128)
// ---------------------------------------------------------------------------
__global__ void __launch_bounds__(128) k_node(
    unsigned short* __restrict__ hbf, float* __restrict__ h,
    const float* __restrict__ aggr,
    const unsigned short* __restrict__ W1, const float* __restrict__ b1,
    const unsigned short* __restrict__ W2, const float* __restrict__ b2, int N) {
  __shared__ unsigned short sact[WPB][16 * 128];
  const int w = threadIdx.x >> 5, lane = threadIdx.x & 31;
  const int tile = blockIdx.x * WPB + w;
  if (tile * 16 >= N) return;
  const int tb = tile * 16, m16 = lane & 15, g = lane >> 4, h8 = g * 8;
  unsigned short* act = sact[w];
  const unsigned short* rowH = hbf + (size_t)(tb + m16) * 128;

  v8f acc[8];
#pragma unroll
  for (int nt = 0; nt < 8; ++nt) {
    float bv = b1[nt * 16 + m16];
#pragma unroll
    for (int r = 0; r < 8; ++r) acc[nt][r] = bv;
  }
#pragma unroll
  for (int ks = 0; ks < 8; ++ks) {
    const int k0 = ks * 32;
    Frag A;
    if (k0 < 128) {
      A = load_a(rowH, k0, g);
    } else {
      const float* ar = aggr + (size_t)(tb + m16) * 128 + (k0 - 128);
      float4 x0 = *reinterpret_cast<const float4*>(ar + h8);
      float4 x1 = *reinterpret_cast<const float4*>(ar + h8 + 4);
      float4 x2 = *reinterpret_cast<const float4*>(ar + 16 + h8);
      float4 x3 = *reinterpret_cast<const float4*>(ar + 16 + h8 + 4);
      A.h[0] = f2bf(x0.x);  A.h[1] = f2bf(x0.y);  A.h[2] = f2bf(x0.z);  A.h[3] = f2bf(x0.w);
      A.h[4] = f2bf(x1.x);  A.h[5] = f2bf(x1.y);  A.h[6] = f2bf(x1.z);  A.h[7] = f2bf(x1.w);
      A.h[8] = f2bf(x2.x);  A.h[9] = f2bf(x2.y);  A.h[10] = f2bf(x2.z); A.h[11] = f2bf(x2.w);
      A.h[12] = f2bf(x3.x); A.h[13] = f2bf(x3.y); A.h[14] = f2bf(x3.z); A.h[15] = f2bf(x3.w);
    }
#pragma unroll
    for (int nt = 0; nt < 8; ++nt) {
      Frag B = load_b(W1, 256, nt, k0, lane);
      acc[nt] = wmma_bf16(A, B, acc[nt]);
    }
  }
#pragma unroll
  for (int nt = 0; nt < 8; ++nt)
#pragma unroll
    for (int r = 0; r < 8; ++r)
      act[(r + h8) * 128 + nt * 16 + m16] = f2bf(fsilu(acc[nt][r]));

#pragma unroll
  for (int nt = 0; nt < 8; ++nt) {
    float bv = b2[nt * 16 + m16];
#pragma unroll
    for (int r = 0; r < 8; ++r) acc[nt][r] = bv;
  }
#pragma unroll
  for (int ks = 0; ks < 4; ++ks) {
    Frag A = load_a(act + m16 * 128, ks * 32, g);
#pragma unroll
    for (int nt = 0; nt < 8; ++nt) {
      Frag B = load_b(W2, 128, nt, ks * 32, lane);
      acc[nt] = wmma_bf16(A, B, acc[nt]);
    }
  }
#pragma unroll
  for (int nt = 0; nt < 8; ++nt)
#pragma unroll
    for (int r = 0; r < 8; ++r) {
      size_t idx = (size_t)(tb + r + h8) * 128 + nt * 16 + m16;
      float hv = h[idx] + fsilu(acc[nt][r]);
      h[idx] = hv;
      hbf[idx] = f2bf(hv);
    }
}

// ---------------------------------------------------------------------------
// Output head: out = tanh(out2(silu(out1(h))))  (out2 is 128->3, VALU)
// ---------------------------------------------------------------------------
__global__ void __launch_bounds__(128) k_out(
    const unsigned short* __restrict__ hbf,
    const unsigned short* __restrict__ W1, const float* __restrict__ b1,
    const float* __restrict__ W2, const float* __restrict__ b2,
    float* __restrict__ out, int N) {
  __shared__ unsigned short sact[WPB][16 * 128];
  const int w = threadIdx.x >> 5, lane = threadIdx.x & 31;
  const int tile = blockIdx.x * WPB + w;
  if (tile * 16 >= N) return;
  const int tb = tile * 16, m16 = lane & 15, g = lane >> 4, h8 = g * 8;
  unsigned short* act = sact[w];
  const unsigned short* rowH = hbf + (size_t)(tb + m16) * 128;

  v8f acc[8];
#pragma unroll
  for (int nt = 0; nt < 8; ++nt) {
    float bv = b1[nt * 16 + m16];
#pragma unroll
    for (int r = 0; r < 8; ++r) acc[nt][r] = bv;
  }
#pragma unroll
  for (int ks = 0; ks < 4; ++ks) {
    Frag A = load_a(rowH, ks * 32, g);
#pragma unroll
    for (int nt = 0; nt < 8; ++nt) {
      Frag B = load_b(W1, 128, nt, ks * 32, lane);
      acc[nt] = wmma_bf16(A, B, acc[nt]);
    }
  }
#pragma unroll
  for (int nt = 0; nt < 8; ++nt)
#pragma unroll
    for (int r = 0; r < 8; ++r)
      act[(r + h8) * 128 + nt * 16 + m16] = f2bf(fsilu(acc[nt][r]));

  // out2: 48 (node, dim) dot products over 32 lanes
  for (int p = lane; p < 48; p += 32) {
    int nl = p / 3, j = p - nl * 3;
    float s = b2[j];
#pragma unroll 4
    for (int k = 0; k < 128; ++k) s += bf2f(act[nl * 128 + k]) * W2[k * 3 + j];
    out[(size_t)(tb + nl) * 3 + j] = tanhf(s);
  }
}

// ---------------------------------------------------------------------------
// Host orchestration
// ---------------------------------------------------------------------------
extern "C" void kernel_launch(void* const* d_in, const int* in_sizes, int n_in,
                              void* d_out, int out_size, void* d_ws, size_t ws_size,
                              hipStream_t stream) {
  const int N = in_sizes[0] / 16;
  const int E = in_sizes[3] / 2;

  const float* x    = (const float*)d_in[0];
  const float* fs   = (const float*)d_in[1];
  const float* fn   = (const float*)d_in[2];
  const long long* eidx = (const long long*)d_in[3];
  const long long* erow = eidx;
  const long long* ecol = eidx + E;
  const float* elen = (const float*)d_in[4];
  const float* evec = (const float*)d_in[5];
  float* outp = (float*)d_out;

  // params pytree order (dict keys sorted): blocks[0..2]{e1,e2,n1,n2,t1,t2}(W,b),
  // emb1(W,b), emb2(W,b), out1(W,b), out2(W,b)
  const int PB = 6;
  auto pf = [&](int i) { return (const float*)d_in[i]; };
  const float* emb1W = pf(PB + 36 + 0); const float* emb1b = pf(PB + 36 + 1);
  const float* emb2W = pf(PB + 36 + 2); const float* emb2b = pf(PB + 36 + 3);
  const float* out1W = pf(PB + 36 + 4); const float* out1b = pf(PB + 36 + 5);
  const float* out2W = pf(PB + 36 + 6); const float* out2b = pf(PB + 36 + 7);

  // workspace bump allocator (256B aligned)
  char* ws = (char*)d_ws;
  size_t off = 0;
  auto alloc = [&](size_t bytes) -> char* {
    char* p = ws + off;
    off = (off + bytes + 255) & ~(size_t)255;
    return p;
  };
  unsigned short* hbf  = (unsigned short*)alloc((size_t)N * 128 * 2);
  float* h    = (float*)alloc((size_t)N * 128 * 4);
  float* aggr = (float*)alloc((size_t)N * 128 * 4);
  float* vsum = (float*)alloc((size_t)N * 3 * 4);
  float* vcnt = (float*)alloc((size_t)N * 4);
  float* ecos = (float*)alloc((size_t)E * 4);
  unsigned short* cWe1[3], *cWe2[3], *cWn1[3], *cWn2[3], *cWt1[3], *cWt2[3];
  for (int b = 0; b < 3; ++b) {
    cWe1[b] = (unsigned short*)alloc((size_t)128 * 320 * 2);
    cWe2[b] = (unsigned short*)alloc((size_t)128 * 128 * 2);
    cWn1[b] = (unsigned short*)alloc((size_t)128 * 256 * 2);
    cWn2[b] = (unsigned short*)alloc((size_t)128 * 128 * 2);
    cWt1[b] = (unsigned short*)alloc((size_t)128 * 136 * 2);
    cWt2[b] = (unsigned short*)alloc((size_t)128 * 128 * 2);
  }
  unsigned short* cEmb1 = (unsigned short*)alloc((size_t)128 * 32 * 2);
  unsigned short* cEmb2 = (unsigned short*)alloc((size_t)128 * 128 * 2);
  unsigned short* cOut1 = (unsigned short*)alloc((size_t)128 * 128 * 2);

  auto cvt = [&](const float* W, unsigned short* Wt, int din, int dout, int stride) {
    int elems = dout * stride;
    k_convert<<<(elems + 255) / 256, 256, 0, stream>>>(W, Wt, din, dout, stride);
  };
  for (int b = 0; b < 3; ++b) {
    int base = PB + b * 12;
    cvt(pf(base + 0),  cWe1[b], 320, 128, 320);
    cvt(pf(base + 2),  cWe2[b], 128, 128, 128);
    cvt(pf(base + 4),  cWn1[b], 256, 128, 256);
    cvt(pf(base + 6),  cWn2[b], 128, 128, 128);
    cvt(pf(base + 8),  cWt1[b], 129, 128, 136);
    cvt(pf(base + 10), cWt2[b], 128, 128, 128);
  }
  cvt(emb1W, cEmb1, 32, 128, 32);
  cvt(emb2W, cEmb2, 128, 128, 128);
  cvt(out1W, cOut1, 128, 128, 128);

  // geometry (block-invariant)
  hipMemsetAsync(vsum, 0, (size_t)N * 3 * 4, stream);
  hipMemsetAsync(vcnt, 0, (size_t)N * 4, stream);
  k_scatter_vec<<<(E + 255) / 256, 256, 0, stream>>>(ecol, evec, vsum, vcnt, E);
  k_div_vec<<<(N + 255) / 256, 256, 0, stream>>>(vsum, vcnt, N);
  k_cos<<<(E + 255) / 256, 256, 0, stream>>>(ecol, evec, elen, vsum, ecos, E);

  const int nTiles = (N + 15) / 16;
  const int eTiles = (E + 15) / 16;
  const dim3 nGrid((nTiles + WPB - 1) / WPB), eGrid((eTiles + WPB - 1) / WPB), blk(128);

  // embedding
  k_embed<<<nGrid, blk, 0, stream>>>(x, fs, fn, cEmb1, emb1b, cEmb2, emb2b, h, hbf, N);

  // 3 message-passing blocks
  for (int b = 0; b < 3; ++b) {
    int base = PB + b * 12;
    hipMemsetAsync(aggr, 0, (size_t)N * 128 * 4, stream);
    k_edge<<<eGrid, blk, 0, stream>>>(hbf, erow, ecol, elen, ecos,
                                      cWe1[b], pf(base + 1), cWe2[b], pf(base + 3),
                                      cWt1[b], pf(base + 9), cWt2[b], pf(base + 11),
                                      aggr, E);
    k_node<<<nGrid, blk, 0, stream>>>(hbf, h, aggr,
                                      cWn1[b], pf(base + 5), cWn2[b], pf(base + 7), N);
  }

  // output head
  k_out<<<nGrid, blk, 0, stream>>>(hbf, cOut1, out1b, out2W, out2b, outp, N);
}
